// ContrastiveLoss_1254130450729
// MI455X (gfx1250) — compile-verified, matrix operand order checked by reading, not run on hardware
//
#include <hip/hip_runtime.h>

typedef __attribute__((ext_vector_type(16))) _Float16 v16h;
typedef __attribute__((ext_vector_type(8)))  _Float16 v8h;
typedef __attribute__((ext_vector_type(8)))  float    v8f;

#define B_ROWS 4096
#define D_DIM  256
#define N_ROWS 8192        // 2B
#define TILE   16
#define KBLK   8           // 256 / 32
#define SLABS  16
#define COLS_PER_SLAB 512
#define CT_PER_SLAB   32   // 512 / 16
#define WAVES_PER_BLK 8
#define ROWS_PER_WAVE 32   // two 16-row WMMA tiles per wave
#define ROWS_PER_BLK  256  // 8 waves * 32 rows
#define TILE_HALFS    (TILE * D_DIM)   // 4096 halfs = 8 KB per B tile

// ---------------------------------------------------------------------------
// Kernel 1: one wave per row-pair i. L2-normalize proj_1[i] and proj_2[i] in
// f32, emit f16 copies into zh[i] and zh[B+i] (WMMA operands), and compute the
// positive pair similarity pos[i] = dot(z1n_i, z2n_i) in full f32.
// ---------------------------------------------------------------------------
__global__ void prep_kernel(const float* __restrict__ p1,
                            const float* __restrict__ p2,
                            _Float16* __restrict__ zh,
                            float* __restrict__ pos) {
    int wave = (blockIdx.x * blockDim.x + threadIdx.x) >> 5;
    int lane = threadIdx.x & 31;
    if (wave >= B_ROWS) return;

    const float* r1 = p1 + (size_t)wave * D_DIM;
    const float* r2 = p2 + (size_t)wave * D_DIM;

    float a[8], b[8];
    float ss1 = 0.f, ss2 = 0.f;
#pragma unroll
    for (int i = 0; i < 8; ++i) {
        a[i] = r1[lane + i * 32];
        b[i] = r2[lane + i * 32];
        ss1 += a[i] * a[i];
        ss2 += b[i] * b[i];
    }
#pragma unroll
    for (int m = 16; m >= 1; m >>= 1) {
        ss1 += __shfl_xor(ss1, m, 32);
        ss2 += __shfl_xor(ss2, m, 32);
    }
    float inv1 = 1.0f / fmaxf(sqrtf(ss1), 1e-12f);
    float inv2 = 1.0f / fmaxf(sqrtf(ss2), 1e-12f);

    _Float16* o1 = zh + (size_t)wave * D_DIM;
    _Float16* o2 = zh + (size_t)(wave + B_ROWS) * D_DIM;
    float dot = 0.f;
#pragma unroll
    for (int i = 0; i < 8; ++i) {
        float x = a[i] * inv1;
        float y = b[i] * inv2;
        dot += x * y;
        o1[lane + i * 32] = (_Float16)x;
        o2[lane + i * 32] = (_Float16)y;
    }
#pragma unroll
    for (int m = 16; m >= 1; m >>= 1) dot += __shfl_xor(dot, m, 32);
    if (lane == 0) {
        pos[wave]          = dot;
        pos[wave + B_ROWS] = dot;
    }
}

// Assemble one 16x32 f16 WMMA operand from two 16-byte fragments.
// CDNA5 16-bit A layout: lane l in [0,15] holds row l with
// K = {k0..k0+7, k0+16..k0+23}; lane l+16 holds row l with
// K = {k0+8..k0+15, k0+24..k0+31}. B (32x16) mirrors this per lane=column.
__device__ __forceinline__ v16h combine16(v8h lo, v8h hi) {
    return __builtin_shufflevector(lo, hi, 0, 1, 2, 3, 4, 5, 6, 7,
                                           8, 9, 10, 11, 12, 13, 14, 15);
}

// ---------------------------------------------------------------------------
// Kernel 2: fused sim-GEMM + exp row-sum.
//   * 8 waves/block, each wave holds TWO 16-row A tiles resident in VGPRs
//     (block covers 256 rows).
//   * The 16-column B tile (16 contiguous zh rows = one linear 8 KB chunk) is
//     staged cooperatively into a double-buffered LDS tile; all 8 waves read
//     their WMMA fragments from LDS (ds_load_b128), so every global B byte
//     feeds 16 output tiles and every LDS fragment feeds 2 WMMAs.
//   * Epilogue fuses exp(2*sim) into per-row partial sums; sim is never
//     materialized.
// ---------------------------------------------------------------------------
__global__ __launch_bounds__(256, 1)
void simsum_kernel(const _Float16* __restrict__ zh,
                   float* __restrict__ denomPart) {
    __shared__ __align__(16) _Float16 lbuf[2][TILE_HALFS];   // 2 x 8 KB

    int widx = threadIdx.x >> 5;        // wave in block: 0..7
    int lane = threadIdx.x & 31;
    int rowBlock = blockIdx.x >> 4;     // 0..31  (256 rows each)
    int slab     = blockIdx.x & 15;     // 0..15  (512 cols each)
    int row0 = rowBlock * ROWS_PER_BLK + widx * ROWS_PER_WAVE;

    int rsel  = lane & 15;              // row-in-tile (A) / col-in-tile (B)
    int khalf = (lane >> 4) * 8;        // half-wave K offset

    // Two resident A tiles: rows [row0, row0+16) and [row0+16, row0+32).
    const _Float16* aptr0 = zh + (size_t)(row0 + rsel) * D_DIM + khalf;
    const _Float16* aptr1 = aptr0 + (size_t)TILE * D_DIM;
    v16h areg0[KBLK], areg1[KBLK];
#pragma unroll
    for (int kb = 0; kb < KBLK; ++kb) {
        const _Float16* p0 = aptr0 + kb * 32;
        const _Float16* p1 = aptr1 + kb * 32;
        areg0[kb] = combine16(*(const v8h*)p0, *(const v8h*)(p0 + 16));
        areg1[kb] = combine16(*(const v8h*)p1, *(const v8h*)(p1 + 16));
    }

    float rowsum0[8], rowsum1[8];
#pragma unroll
    for (int g = 0; g < 8; ++g) { rowsum0[g] = 0.f; rowsum1[g] = 0.f; }

    int col0base = slab * COLS_PER_SLAB;
    int t8 = threadIdx.x * 8;           // this thread's 8-half staging slice

    // Prologue: stage tile 0 into buffer 0 (B tile is one contiguous 8 KB).
    {
        const _Float16* src = zh + (size_t)col0base * D_DIM;
        *(v8h*)(&lbuf[0][t8])        = *(const v8h*)(src + t8);
        *(v8h*)(&lbuf[0][2048 + t8]) = *(const v8h*)(src + 2048 + t8);
    }
    __syncthreads();

    for (int ct = 0; ct < CT_PER_SLAB; ++ct) {
        int cur = ct & 1;
        int nxt = cur ^ 1;

        // Issue next tile's global loads now; their wait lands after compute.
        v8h t0, t1;
        bool have_next = (ct + 1) < CT_PER_SLAB;
        if (have_next) {
            const _Float16* src = zh + (size_t)(col0base + (ct + 1) * TILE) * D_DIM;
            t0 = *(const v8h*)(src + t8);
            t1 = *(const v8h*)(src + 2048 + t8);
        }

        // Compute: 8 K-steps, each B fragment from LDS feeds two WMMAs.
        const _Float16* bbase = &lbuf[cur][(size_t)rsel * D_DIM + khalf];
        v8f acc0 = {0.f, 0.f, 0.f, 0.f, 0.f, 0.f, 0.f, 0.f};
        v8f acc1 = {0.f, 0.f, 0.f, 0.f, 0.f, 0.f, 0.f, 0.f};
#pragma unroll
        for (int kb = 0; kb < KBLK; ++kb) {
            const _Float16* bp = bbase + kb * 32;
            v16h breg = combine16(*(const v8h*)bp, *(const v8h*)(bp + 16));
            acc0 = __builtin_amdgcn_wmma_f32_16x16x32_f16(
                false, areg0[kb], false, breg, (short)0, acc0, false, false);
            acc1 = __builtin_amdgcn_wmma_f32_16x16x32_f16(
                false, areg1[kb], false, breg, (short)0, acc1, false, false);
        }

        // exp(sim/T), T = 0.5 -> exp(2*sim); trans pipe co-executes with WMMA.
        // C/D layout: VGPR g of lane l holds row (g + 8*(l>>4)), col (l&15).
#pragma unroll
        for (int g = 0; g < 8; ++g) {
            rowsum0[g] += __expf(2.0f * acc0[g]);
            rowsum1[g] += __expf(2.0f * acc1[g]);
        }

        // Commit next tile into the other LDS buffer, then block barrier.
        if (have_next) {
            *(v8h*)(&lbuf[nxt][t8])        = t0;
            *(v8h*)(&lbuf[nxt][2048 + t8]) = t1;
        }
        __syncthreads();
    }

    // Reduce across the 16 columns held by each half-wave (xor over low 4 bits).
#pragma unroll
    for (int m = 1; m <= 8; m <<= 1) {
#pragma unroll
        for (int g = 0; g < 8; ++g) {
            rowsum0[g] += __shfl_xor(rowsum0[g], m, 32);
            rowsum1[g] += __shfl_xor(rowsum1[g], m, 32);
        }
    }
    if (rsel == 0) {
        int rbase = row0 + (lane >> 4) * 8;  // lane0 -> rows 0..7, lane16 -> 8..15
        float* outp = denomPart + (size_t)slab * N_ROWS + rbase;
#pragma unroll
        for (int g = 0; g < 8; ++g) {
            outp[g]        = rowsum0[g];
            outp[g + TILE] = rowsum1[g];
        }
    }
}

// ---------------------------------------------------------------------------
// Kernel 3: fold slab partials, subtract the diagonal term exp(2*1), combine
// with positives:  loss = mean( -2*pos_i + log(denom_i) ).
// ---------------------------------------------------------------------------
__global__ void loss_kernel(const float* __restrict__ pos,
                            const float* __restrict__ denomPart,
                            float* __restrict__ out) {
    __shared__ float red[256];
    const float e2 = __expf(2.0f);   // exp(sim_ii / T), sim_ii == 1 after normalize
    float acc = 0.f;
    for (int i = threadIdx.x; i < N_ROWS; i += 256) {
        float d = 0.f;
#pragma unroll
        for (int s = 0; s < SLABS; ++s) d += denomPart[(size_t)s * N_ROWS + i];
        d -= e2;
        acc += -2.0f * pos[i] + __logf(d);
    }
    red[threadIdx.x] = acc;
    __syncthreads();
    for (int s = 128; s > 0; s >>= 1) {
        if ((int)threadIdx.x < s) red[threadIdx.x] += red[threadIdx.x + s];
        __syncthreads();
    }
    if (threadIdx.x == 0) out[0] = red[0] / (float)N_ROWS;
}

extern "C" void kernel_launch(void* const* d_in, const int* in_sizes, int n_in,
                              void* d_out, int out_size, void* d_ws, size_t ws_size,
                              hipStream_t stream) {
    const float* p1 = (const float*)d_in[0];
    const float* p2 = (const float*)d_in[1];

    char* ws = (char*)d_ws;
    _Float16* zh     = (_Float16*)ws;                                  // 4 MB
    float* pos       = (float*)(ws + (size_t)N_ROWS * D_DIM * sizeof(_Float16));
    float* denomPart = pos + N_ROWS;                                   // 16 * 8192 f32
    float* out = (float*)d_out;

    // 4096 waves (one per row pair), 8 waves per 256-thread block.
    prep_kernel<<<512, 256, 0, stream>>>(p1, p2, zh, pos);
    // 32 row-blocks * 16 column slabs = 512 blocks of 8 waves.
    simsum_kernel<<<512, 256, 0, stream>>>(zh, denomPart);
    loss_kernel<<<1, 256, 0, stream>>>(pos, denomPart, out);
}
